// ModalCrossAttention_30691836297754
// MI455X (gfx1250) — matrix operand order repaired and checked by reading, not compiled, not run on hardware
//
#include <hip/hip_runtime.h>
#include <stdint.h>

// ---------------------------------------------------------------------------
// ModalCrossAttention on gfx1250 (MI455X): bf16 WMMA pipeline, round 2
//   - one-time f32->bf16 convert (+ transpose) of activations & weights
//   - all GEMM fragments loaded as b128 vectors (no converts in hot loops)
//   - flash attention streams K/V^T chunks via global_load_async_to_lds_b128
//     (double buffered, s_wait_asynccnt) and consumes them with ds_load_b128
// ---------------------------------------------------------------------------

typedef __attribute__((ext_vector_type(16))) __bf16 v16bf;
typedef __attribute__((ext_vector_type(8)))  __bf16 v8bf;
typedef __attribute__((ext_vector_type(8)))  float  v8f;

#define BATCH 2
#define CIN   256
#define HID   128
#define NHEAD 8
#define HDIM  16
#define NPIX  2304          // 48*48
#define NT    (NPIX / 16)   // 144 key/query tiles
#define NITER (NT / 2)      // 72 iterations of 32 keys
#define ATT_SCALE 0.25f     // 16^-0.5

__device__ __forceinline__ __bf16 f2bf(float f) {
  uint32_t u = __builtin_bit_cast(uint32_t, f);
  uint32_t r = (u + 0x7FFFu + ((u >> 16) & 1u)) >> 16;   // round-to-nearest-even
  return __builtin_bit_cast(__bf16, (uint16_t)r);
}

__device__ __forceinline__ v8f wmma_bf16(v16bf a, v16bf b, v8f c) {
  return __builtin_amdgcn_wmma_f32_16x16x32_bf16(
      false, a, false, b, (short)0, c, false, false);
}

// gfx1250 async copy: global -> LDS, 16 bytes per lane (ISA 10, ASYNCcnt)
__device__ __forceinline__ void async_load_b128(uint32_t lds_byte_off,
                                                const void* gaddr) {
  asm volatile("global_load_async_to_lds_b128 %0, %1, off"
               :: "v"(lds_byte_off), "v"(gaddr) : "memory");
}

// ---------------------------------------------------------------------------
// One-time converts: activations f32 [B][C][N] -> bf16 [B][N][C] (transposed),
// weights f32 [O][C] -> bf16 [O][C]
// ---------------------------------------------------------------------------
__global__ __launch_bounds__(256)
void xpose_bf16_kernel(const float* __restrict__ x, __bf16* __restrict__ xt) {
  const int i = blockIdx.x * 256 + threadIdx.x;     // n fastest -> coalesced read
  const int n = i % NPIX;
  const int c = (i / NPIX) % CIN;
  const int b = i / (NPIX * CIN);
  xt[((size_t)b * NPIX + n) * CIN + c] = f2bf(x[i]);
}

__global__ __launch_bounds__(256)
void cvt_bf16_kernel(const float* __restrict__ in, __bf16* __restrict__ out, int n) {
  const int i = blockIdx.x * 256 + threadIdx.x;
  if (i < n) out[i] = f2bf(in[i]);
}

// ---------------------------------------------------------------------------
// Kernel 1: QKV projection GEMM (bf16 in, bf16 out).
//   Xt: [B][N][CIN] bf16, W: [128][CIN] bf16, bias f32.
//   Q/K out: [b][h][n][d]; V out transposed: [b][h][d][n].
// grid (NT, 3*8, B), block 32
// ---------------------------------------------------------------------------
__global__ __launch_bounds__(32)
void qkv_proj_kernel(const __bf16* __restrict__ xt,
                     const __bf16* __restrict__ wq, const float* __restrict__ bq,
                     const __bf16* __restrict__ wk, const float* __restrict__ bk,
                     const __bf16* __restrict__ wv, const float* __restrict__ bv,
                     __bf16* __restrict__ outq, __bf16* __restrict__ outk,
                     __bf16* __restrict__ outv)
{
  const int lane = threadIdx.x & 31;
  const int m  = lane & 15;
  const int hi = lane >> 4;
  const int pt = blockIdx.x;
  const int p  = blockIdx.y >> 3;     // 0=q 1=k 2=v
  const int h  = blockIdx.y & 7;
  const int b  = blockIdx.z;
  const int n0 = pt * 16;
  const int o0 = h * 16;

  const __bf16* w  = (p == 0) ? wq : (p == 1) ? wk : wv;
  const float*  bs = (p == 0) ? bq : (p == 1) ? bk : bv;

  const __bf16* xrow = xt + ((size_t)b * NPIX + n0 + m) * CIN;   // A row (pixel)
  const __bf16* wrow = w + (size_t)(o0 + m) * CIN;               // B col (out ch)

  v8f acc = {};
  #pragma unroll
  for (int c0 = 0; c0 < CIN; c0 += 32) {
    // A fragment: lane row m; K = hi*16 .. hi*16+15
    v8bf a_lo = *(const v8bf*)(xrow + c0 + hi * 16);
    v8bf a_hi = *(const v8bf*)(xrow + c0 + hi * 16 + 8);
    v16bf A;
    #pragma unroll
    for (int e = 0; e < 8; ++e) { A[e] = a_lo[e]; A[e + 8] = a_hi[e]; }
    // B fragment: lane col m; K = hi*16 .. hi*16+15 (contiguous 32B)
    v16bf Bf = *(const v16bf*)(wrow + c0 + hi * 16);
    acc = wmma_bf16(A, Bf, acc);
  }

  const float bb = bs[o0 + m];
  if (p == 2) {                       // V: store transposed [b][h][d][n]
    __bf16* ob = outv + ((size_t)b * NHEAD + h) * NPIX * HDIM;
    #pragma unroll
    for (int r = 0; r < 8; ++r)
      ob[(size_t)m * NPIX + n0 + hi * 8 + r] = f2bf(acc[r] + bb);
  } else {                            // Q (scaled) / K: [b][h][n][d]
    const float scale = (p == 0) ? ATT_SCALE : 1.0f;
    __bf16* ob = ((p == 0) ? outq : outk) + ((size_t)b * NHEAD + h) * NPIX * HDIM;
    #pragma unroll
    for (int r = 0; r < 8; ++r)
      ob[(size_t)(n0 + hi * 8 + r) * HDIM + m] = f2bf((acc[r] + bb) * scale);
  }
}

// ---------------------------------------------------------------------------
// Kernel 2: flash cross-attention, one wave per (dir, b, head, q-tile).
// K and V^T chunks (1 KB each) double-buffered in LDS via async loads.
// grid (NT, NHEAD, 2*B), block 32
// ---------------------------------------------------------------------------
__global__ __launch_bounds__(32)
void cross_attn_kernel(const __bf16* __restrict__ Q1, const __bf16* __restrict__ K1,
                       const __bf16* __restrict__ V1t,
                       const __bf16* __restrict__ Q2, const __bf16* __restrict__ K2,
                       const __bf16* __restrict__ V2t,
                       __bf16* __restrict__ O1, __bf16* __restrict__ O2)
{
  __shared__ __align__(32) uint16_t ldsK[2][512];   // [buf][32 keys * 16 d]
  __shared__ __align__(32) uint16_t ldsV[2][512];   // [buf][16 d * 32 keys]
  __shared__ __align__(32) uint16_t ldsP[512];      // P tile 16x32 (transpose stage)

  const int lane = threadIdx.x & 31;
  const int m  = lane & 15;
  const int hi = lane >> 4;
  const int qt  = blockIdx.x;
  const int h   = blockIdx.y;
  const int b   = blockIdx.z % BATCH;
  const int dir = blockIdx.z / BATCH;

  const __bf16* Q  = dir ? Q2 : Q1;
  const __bf16* K  = dir ? K1 : K2;
  const __bf16* Vt = dir ? V1t : V2t;
  __bf16*       O  = dir ? O2 : O1;

  const size_t hoff = ((size_t)b * NHEAD + h) * NPIX * HDIM;
  const __bf16* Qh  = Q + hoff;
  const __bf16* Kh  = K + hoff;
  const __bf16* Vth = Vt + hoff;     // [d][n]
  __bf16*       Oh  = O + hoff;

  const int n0 = qt * 16;

  // Q tile A-fragment (head_dim 16 zero-padded to K=32)
  v16bf qv = *(const v16bf*)(Qh + (size_t)(n0 + m) * HDIM);
  v16bf aq;
  #pragma unroll
  for (int e = 0; e < 16; ++e) {
    aq[e] = hi ? __builtin_bit_cast(__bf16, (uint16_t)0) : qv[e];
  }

  // async prefetch of one 32-key chunk (K: 1KB contiguous; Vt: 16 rows x 64B)
  auto prefetch = [&](int it, int t) {
    const int j0 = it * 32;
    const uint32_t kb = (uint32_t)(uintptr_t)&ldsK[t][0];
    const uint32_t vb = (uint32_t)(uintptr_t)&ldsV[t][0];
    const char* gk = (const char*)(Kh + (size_t)j0 * HDIM);
    async_load_b128(kb + lane * 16, gk + lane * 16);
    async_load_b128(kb + 512 + lane * 16, gk + 512 + lane * 16);
    const int s0 = lane, s1 = lane + 32;          // 64 x 16B segments
    async_load_b128(vb + s0 * 16,
                    Vth + (size_t)(s0 >> 2) * NPIX + j0 + (s0 & 3) * 8);
    async_load_b128(vb + s1 * 16,
                    Vth + (size_t)(s1 >> 2) * NPIX + j0 + (s1 & 3) * 8);
  };

  float mrun[8], lrun[8];
  v8f oacc = {};
  #pragma unroll
  for (int r = 0; r < 8; ++r) { mrun[r] = -1e30f; lrun[r] = 0.0f; }

  prefetch(0, 0);

  for (int it = 0; it < NITER; ++it) {
    if (it + 1 < NITER) {
      prefetch(it + 1, (it + 1) & 1);
      asm volatile("s_wait_asynccnt 4" ::: "memory");   // chunk `it` resident
    } else {
      asm volatile("s_wait_asynccnt 0" ::: "memory");
    }
    const int t = it & 1;

    // S = Q K^T for two 16-key tiles; B-fragment cols = key pixels (from LDS)
    const v16bf zv = {};
    v16bf bk0 = hi ? zv : *(const v16bf*)&ldsK[t][m * 16];
    v16bf bk1 = hi ? zv : *(const v16bf*)&ldsK[t][(16 + m) * 16];
    v8f zc = {};
    v8f s0 = wmma_bf16(aq, bk0, zc);
    v8f s1 = wmma_bf16(aq, bk1, zc);

    // online softmax (rows live across each 16-lane half-wave group)
    float p0[8], p1[8];
    #pragma unroll
    for (int r = 0; r < 8; ++r) {
      float tmax = fmaxf(s0[r], s1[r]);
      #pragma unroll
      for (int off = 8; off >= 1; off >>= 1)
        tmax = fmaxf(tmax, __shfl_xor(tmax, off, 32));
      const float mnew = fmaxf(mrun[r], tmax);
      const float corr = __expf(mrun[r] - mnew);
      p0[r] = __expf(s0[r] - mnew);
      p1[r] = __expf(s1[r] - mnew);
      float rs = p0[r] + p1[r];
      #pragma unroll
      for (int off = 8; off >= 1; off >>= 1)
        rs += __shfl_xor(rs, off, 32);
      lrun[r] = lrun[r] * corr + rs;
      oacc[r] = oacc[r] * corr;
      mrun[r] = mnew;
    }

    // transpose P (C-layout regs) -> A-fragment layout via LDS
    #pragma unroll
    for (int r = 0; r < 8; ++r) {
      const int row = hi * 8 + r;
      ldsP[row * 32 + m]      = __builtin_bit_cast(uint16_t, f2bf(p0[r]));
      ldsP[row * 32 + 16 + m] = __builtin_bit_cast(uint16_t, f2bf(p1[r]));
    }
    __syncthreads();

    v8bf p_lo = *(const v8bf*)&ldsP[m * 32 + hi * 16];
    v8bf p_hi = *(const v8bf*)&ldsP[m * 32 + hi * 16 + 8];
    v16bf ap;
    #pragma unroll
    for (int e = 0; e < 8; ++e) { ap[e] = p_lo[e]; ap[e + 8] = p_hi[e]; }
    // V^T chunk as B-fragment: lane col = d, K = 32 key pixels (contiguous 32B)
    v16bf bv = *(const v16bf*)&ldsV[t][m * 32 + hi * 16];
    oacc = wmma_bf16(ap, bv, oacc);
    __syncthreads();
  }

  #pragma unroll
  for (int r = 0; r < 8; ++r)
    Oh[(size_t)(n0 + hi * 8 + r) * HDIM + m] = f2bf(oacc[r] / lrun[r]);
}

// ---------------------------------------------------------------------------
// Kernel 3: output projection. O [b][h][n][d] bf16, W_o [256][128] bf16.
// out f32 [mod][b][oc][n] (tuple concatenated). grid (NT, 16, 2*B), block 32
// ---------------------------------------------------------------------------
__global__ __launch_bounds__(32)
void out_proj_kernel(const __bf16* __restrict__ O1, const __bf16* __restrict__ O2,
                     const __bf16* __restrict__ wo1, const float* __restrict__ bo1,
                     const __bf16* __restrict__ wo2, const float* __restrict__ bo2,
                     float* __restrict__ out)
{
  const int lane = threadIdx.x & 31;
  const int m  = lane & 15;
  const int hi = lane >> 4;
  const int pt  = blockIdx.x;
  const int ot  = blockIdx.y;
  const int b   = blockIdx.z % BATCH;
  const int mod = blockIdx.z / BATCH;

  const __bf16* Ox = mod ? O2 : O1;
  const __bf16* w  = mod ? wo2 : wo1;
  const float*  bs = mod ? bo2 : bo1;
  float* outp = out + (size_t)mod * BATCH * CIN * NPIX;

  const int n0 = pt * 16;
  const int o0 = ot * 16;
  const __bf16* Ob   = Ox + (size_t)b * NHEAD * NPIX * HDIM;
  const __bf16* wrow = w + (size_t)(o0 + m) * HID;

  v8f acc = {};
  #pragma unroll
  for (int c0 = 0; c0 < HID; c0 += 32) {
    const int h0 = c0 >> 4;                        // hidden = h*16 + d
    v8bf a_lo = *(const v8bf*)(Ob + ((size_t)(h0 + hi) * NPIX + n0 + m) * HDIM);
    v8bf a_hi = *(const v8bf*)(Ob + ((size_t)(h0 + hi) * NPIX + n0 + m) * HDIM + 8);
    v16bf A;
    #pragma unroll
    for (int e = 0; e < 8; ++e) { A[e] = a_lo[e]; A[e + 8] = a_hi[e]; }
    v16bf Bf = *(const v16bf*)(wrow + c0 + hi * 16);
    acc = wmma_bf16(A, Bf, acc);
  }

  const float bb = bs[o0 + m];
  #pragma unroll
  for (int r = 0; r < 8; ++r)
    outp[((size_t)b * CIN + o0 + m) * NPIX + n0 + hi * 8 + r] = acc[r] + bb;
}

// ---------------------------------------------------------------------------
extern "C" void kernel_launch(void* const* d_in, const int* in_sizes, int n_in,
                              void* d_out, int out_size, void* d_ws, size_t ws_size,
                              hipStream_t stream)
{
  (void)in_sizes; (void)n_in; (void)out_size; (void)ws_size;

  const float* mod1 = (const float*)d_in[0];
  const float* mod2 = (const float*)d_in[1];
  const float* fw[8] = { (const float*)d_in[2],  (const float*)d_in[4],
                         (const float*)d_in[6],  (const float*)d_in[8],
                         (const float*)d_in[10], (const float*)d_in[12],
                         (const float*)d_in[14], (const float*)d_in[16] };
  const float* b_q1 = (const float*)d_in[3];  const float* b_k1 = (const float*)d_in[5];
  const float* b_v1 = (const float*)d_in[7];  const float* b_q2 = (const float*)d_in[9];
  const float* b_k2 = (const float*)d_in[11]; const float* b_v2 = (const float*)d_in[13];
  const float* b_o1 = (const float*)d_in[15]; const float* b_o2 = (const float*)d_in[17];

  // workspace layout (bf16 elements)
  const size_t XTSZ = (size_t)BATCH * NPIX * CIN;        // 1,179,648
  const size_t WSZ  = (size_t)HID * CIN;                 //    32,768
  const size_t HSZ  = (size_t)BATCH * NHEAD * NPIX * HDIM; // 589,824
  __bf16* ws  = (__bf16*)d_ws;
  __bf16* Xt1 = ws;
  __bf16* Xt2 = Xt1 + XTSZ;
  __bf16* wb[8];                                         // q1,k1,v1,q2,k2,v2,o1,o2
  for (int i = 0; i < 8; ++i) wb[i] = Xt2 + XTSZ + (size_t)i * WSZ;
  __bf16* Q1  = wb[7] + WSZ;
  __bf16* K1  = Q1 + HSZ;  __bf16* V1t = K1 + HSZ;
  __bf16* Q2  = V1t + HSZ; __bf16* K2  = Q2 + HSZ;  __bf16* V2t = K2 + HSZ;
  __bf16* O1  = V2t + HSZ; __bf16* O2  = O1 + HSZ;

  const dim3 blk32(32, 1, 1), blk256(256, 1, 1);
  const int xblocks = (int)((size_t)BATCH * CIN * NPIX / 256);

  xpose_bf16_kernel<<<dim3(xblocks), blk256, 0, stream>>>(mod1, Xt1);
  xpose_bf16_kernel<<<dim3(xblocks), blk256, 0, stream>>>(mod2, Xt2);
  for (int i = 0; i < 8; ++i)
    cvt_bf16_kernel<<<dim3((int)(WSZ / 256)), blk256, 0, stream>>>(fw[i], wb[i], (int)WSZ);

  qkv_proj_kernel<<<dim3(NT, 24, BATCH), blk32, 0, stream>>>(
      Xt1, wb[0], b_q1, wb[1], b_k1, wb[2], b_v1, Q1, K1, V1t);
  qkv_proj_kernel<<<dim3(NT, 24, BATCH), blk32, 0, stream>>>(
      Xt2, wb[3], b_q2, wb[4], b_k2, wb[5], b_v2, Q2, K2, V2t);
  cross_attn_kernel<<<dim3(NT, NHEAD, 2 * BATCH), blk32, 0, stream>>>(
      Q1, K1, V1t, Q2, K2, V2t, O1, O2);
  out_proj_kernel<<<dim3(NT, CIN / 16, 2 * BATCH), blk32, 0, stream>>>(
      O1, O2, wb[6], b_o1, wb[7], b_o2, (float*)d_out);
}
